// MultiheadDiffAttn_36137854829100
// MI455X (gfx1250) — matrix operand order repaired and checked by reading, not compiled, not run on hardware
//
#include <hip/hip_runtime.h>

#define SZ 2048          // S == E == 2048
#define HDm 64           // differential head dim

typedef unsigned short u16;
typedef unsigned int   u32;
typedef __attribute__((ext_vector_type(16))) __bf16 v16bf;
typedef __attribute__((ext_vector_type(8)))  float  v8f;
typedef __attribute__((ext_vector_type(16))) u16    v16u;
typedef __attribute__((ext_vector_type(8)))  u16    v8u;

static __device__ __forceinline__ u16 f2bf(float f) {
  u32 u = __float_as_uint(f);
  u32 r = u + 0x7FFFu + ((u >> 16) & 1u);   // round-to-nearest-even
  return (u16)(r >> 16);
}

static __device__ __forceinline__ v8f zero8() {
  v8f z;
#pragma unroll
  for (int i = 0; i < 8; i++) z[i] = 0.f;
  return z;
}

// Build a 16-elem bf16 A/B fragment from two 16-byte chunks.
static __device__ __forceinline__ v16bf frag16(v8u lo, v8u hi) {
  v16u v = __builtin_shufflevector(lo, hi, 0,1,2,3,4,5,6,7,8,9,10,11,12,13,14,15);
  return __builtin_bit_cast(v16bf, v);
}
static __device__ __forceinline__ v16bf loadB16(const u16* p) {
  v16u v = *(const v16u*)p;
  return __builtin_bit_cast(v16bf, v);
}

static __device__ __forceinline__ v8f wmma_bf16(v16bf a, v16bf b, v8f c) {
  // D = A(16x32) * B(32x16) + C, f32 accumulate
  return __builtin_amdgcn_wmma_f32_16x16x32_bf16(false, a, false, b, (short)0, c, false, false);
}

// CDNA5 async copy: per lane, 16B global -> LDS, tracked by ASYNCcnt.
// lds_off = 32-bit LDS byte address (low 32 bits of generic pointer), src = global.
static __device__ __forceinline__ void async_cp16(u32 lds_off, const u16* src) {
  asm volatile("global_load_async_to_lds_b128 %0, %1, off"
               :: "v"(lds_off), "v"(src) : "memory");
}

// ---------------------------------------------------------------- f32 -> bf16
__global__ void cvt_f32_bf16(const float* __restrict__ in, u16* __restrict__ out, int n) {
  int i = blockIdx.x * blockDim.x + threadIdx.x;
  int stride = gridDim.x * blockDim.x;
  for (; i < n; i += stride) out[i] = f2bf(in[i]);
}

// ------------------------------------------------- bf16 transpose (for V^T)
__global__ void transpose_bf16(const u16* __restrict__ in, u16* __restrict__ out) {
  __shared__ u16 tile[32][33];
  int x  = blockIdx.x * 32 + threadIdx.x;
  int y0 = blockIdx.y * 32;
  for (int r = threadIdx.y; r < 32; r += 8)
    tile[r][threadIdx.x] = in[(size_t)(y0 + r) * SZ + x];
  __syncthreads();
  int xo = blockIdx.y * 32 + threadIdx.x;
  for (int r = threadIdx.y; r < 32; r += 8)
    out[(size_t)(blockIdx.x * 32 + r) * SZ + xo] = tile[threadIdx.x][r];
}

// --------------------------------------------------------- bf16 WMMA GEMM
// C(M,N) = A(M,K) * B(N,K)^T  (B stored N-major, K contiguous).
// 64x64 tile per wave, 4 waves/block. Each wave async-stages its own 64x32
// A/B sub-tiles into double-buffered LDS (ASYNCcnt-pipelined, no barriers),
// consumes fragments via ds_load_b128, 16 WMMAs per 32-deep k-step.
template<bool F32OUT>
__global__ __launch_bounds__(128)
void gemm_wmma(const u16* __restrict__ A, const u16* __restrict__ B,
               u16* __restrict__ Cb, float* __restrict__ Cf, float scale) {
  __shared__ u16 sA[4][2][64 * 32];
  __shared__ u16 sB[4][2][64 * 32];
  const int tid  = threadIdx.x;
  const int lane = tid & 31, wave = tid >> 5;
  const int half = lane >> 4, l16 = lane & 15;
  const int m0 = blockIdx.y * 128 + (wave >> 1) * 64;
  const int n0 = blockIdx.x * 128 + (wave & 1) * 64;

  const u32 aBase[2] = { (u32)(uintptr_t)&sA[wave][0][0], (u32)(uintptr_t)&sA[wave][1][0] };
  const u32 bBase[2] = { (u32)(uintptr_t)&sB[wave][0][0], (u32)(uintptr_t)&sB[wave][1][0] };

  auto stage = [&](int buf, int k0) {           // 16 async b128 per wave
#pragma unroll
    for (int rr = 0; rr < 2; rr++) {
      const int row = rr * 32 + lane;           // 64 rows / 32 lanes
      const u16* ga = A + (size_t)(m0 + row) * SZ + k0;
      const u16* gb = B + (size_t)(n0 + row) * SZ + k0;
#pragma unroll
      for (int c = 0; c < 4; c++) {
        async_cp16(aBase[buf] + (u32)(row * 32 + c * 8) * 2, ga + c * 8);
        async_cp16(bBase[buf] + (u32)(row * 32 + c * 8) * 2, gb + c * 8);
      }
    }
  };

  v8f acc[4][4];
#pragma unroll
  for (int i = 0; i < 4; i++)
#pragma unroll
    for (int j = 0; j < 4; j++) acc[i][j] = zero8();

  stage(0, 0);
  for (int k0 = 0, it = 0; k0 < SZ; k0 += 32, ++it) {
    const int cur = it & 1;
    if (k0 + 32 < SZ) {
      stage(cur ^ 1, k0 + 32);                       // prefetch next tile
      asm volatile("s_wait_asynccnt 0x10" ::: "memory");  // prev 16 done, 16 in flight
    } else {
      asm volatile("s_wait_asynccnt 0x0" ::: "memory");   // drain
    }

    v16bf af[4], bfg[4];
#pragma unroll
    for (int i = 0; i < 4; i++) {
      const u16* p = &sA[wave][cur][(i * 16 + l16) * 32 + half * 16];
      af[i] = loadB16(p);
    }
#pragma unroll
    for (int j = 0; j < 4; j++)
      bfg[j] = loadB16(&sB[wave][cur][(j * 16 + l16) * 32 + half * 16]);

#pragma unroll
    for (int i = 0; i < 4; i++)
#pragma unroll
      for (int j = 0; j < 4; j++)
        acc[i][j] = wmma_bf16(af[i], bfg[j], acc[i][j]);
  }

#pragma unroll
  for (int i = 0; i < 4; i++)
#pragma unroll
    for (int j = 0; j < 4; j++)
#pragma unroll
      for (int r = 0; r < 8; r++) {
        int row = m0 + i * 16 + r + half * 8;
        int col = n0 + j * 16 + l16;
        float v = acc[i][j][r] * scale;
        if constexpr (F32OUT) Cf[(size_t)row * SZ + col] = v;
        else                  Cb[(size_t)row * SZ + col] = f2bf(v);
      }
}

// ------------------------------------------------ flash diff-attention core
// One wave: 16 query rows of one sub-head (h2 in 0..31). Online softmax over
// 2048 keys in blocks of 32; P@V accumulated into 16x128 f32.
__global__ __launch_bounds__(128)
void diffattn_fwd(const u16* __restrict__ Qb, const u16* __restrict__ Kb,
                  const u16* __restrict__ Vt, float* __restrict__ Osub) {
  __shared__ float sS[4][16 * 32];
  __shared__ u16   sP[4][16 * 32];
  const int lane = threadIdx.x & 31;
  const int wave = threadIdx.x >> 5;
  const int half = lane >> 4;
  const int l16  = lane & 15;
  const int gw = blockIdx.x * 4 + wave;
  const int h2 = gw >> 7;        // sub-head 0..31
  const int qt = gw & 127;       // q-tile 0..127
  const int q0 = qt * 16;
  float* myS = sS[wave];
  u16*   myP = sP[wave];

  // persistent Q fragments: 16x64 -> two 16x32 A-frags (scaling folded in)
  v16bf aq[2];
#pragma unroll
  for (int kk = 0; kk < 2; kk++) {
    const u16* p = Qb + (size_t)(q0 + l16) * SZ + h2 * HDm + kk * 32 + half * 16;
    aq[kk] = loadB16(p);
  }

  v8f o[8];
#pragma unroll
  for (int nv = 0; nv < 8; nv++) o[nv] = zero8();
  float m_run = -1e30f, l_run = 0.f;

  for (int j0 = 0; j0 < SZ; j0 += 32) {
    // ---- scores: 16 q-rows x 32 keys = two 16x16 C tiles, K=64 over hd
    v8f c0 = zero8(), c1 = zero8();
#pragma unroll
    for (int kk = 0; kk < 2; kk++) {
      const u16* pk0 = Kb + (size_t)(j0 + l16) * SZ      + h2 * HDm + kk * 32 + half * 16;
      const u16* pk1 = Kb + (size_t)(j0 + 16 + l16) * SZ + h2 * HDm + kk * 32 + half * 16;
      c0 = wmma_bf16(aq[kk], loadB16(pk0), c0);
      c1 = wmma_bf16(aq[kk], loadB16(pk1), c1);
    }
    // ---- spill scores to LDS (row-major 16x32)
#pragma unroll
    for (int r = 0; r < 8; r++) {
      int row = r + half * 8;
      myS[row * 32 + l16]      = c0[r];
      myS[row * 32 + 16 + l16] = c1[r];
    }
    // ---- per-row stats: lane owns row l16, columns half*16..+15
    float x[16];
#pragma unroll
    for (int i = 0; i < 16; i++) x[i] = myS[l16 * 32 + half * 16 + i];
    float mx = x[0];
#pragma unroll
    for (int i = 1; i < 16; i++) mx = fmaxf(mx, x[i]);
    mx = fmaxf(mx, __shfl_xor(mx, 16, 32));
    float mnew = fmaxf(m_run, mx);
    float ssum = 0.f;
#pragma unroll
    for (int i = 0; i < 16; i++) { x[i] = __expf(x[i] - mnew); ssum += x[i]; }
    ssum += __shfl_xor(ssum, 16, 32);
    float alpha = __expf(m_run - mnew);
    l_run = l_run * alpha + ssum;
    m_run = mnew;
    // ---- rescale running O by per-row alpha
    float ar[8];
#pragma unroll
    for (int r = 0; r < 8; r++) ar[r] = __shfl(alpha, r + half * 8, 32);
#pragma unroll
    for (int nv = 0; nv < 8; nv++)
#pragma unroll
      for (int r = 0; r < 8; r++) o[nv][r] *= ar[r];
    // ---- P -> bf16 into LDS (row-major)
    u32* pp = (u32*)&myP[l16 * 32 + half * 16];
#pragma unroll
    for (int i = 0; i < 8; i++)
      pp[i] = (u32)f2bf(x[2 * i]) | ((u32)f2bf(x[2 * i + 1]) << 16);
    // ---- reload P as A-fragment (16x32)
    const u16* pa = &myP[l16 * 32 + half * 16];
    v16bf pfrag = loadB16(pa);
    // ---- P @ V: V^T rows are dims (contiguous keys) -> direct B-frags
#pragma unroll
    for (int nv = 0; nv < 8; nv++) {
      const u16* pv = Vt + (size_t)((h2 >> 1) * 128 + nv * 16 + l16) * SZ + j0 + half * 16;
      o[nv] = wmma_bf16(pfrag, loadB16(pv), o[nv]);
    }
  }

  // ---- finalize: divide by l, store f32
  float inv = 1.f / l_run;
  float ir[8];
#pragma unroll
  for (int r = 0; r < 8; r++) ir[r] = __shfl(inv, r + half * 8, 32);
  float* outp = Osub + (size_t)h2 * SZ * 128;
#pragma unroll
  for (int nv = 0; nv < 8; nv++)
#pragma unroll
    for (int r = 0; r < 8; r++)
      outp[(size_t)(q0 + r + half * 8) * 128 + nv * 16 + l16] = o[nv][r] * ir[r];
}

// ----------------------------------------------- lambda_full scalar (1 block)
__global__ void compute_lambda(const float* __restrict__ lq1, const float* __restrict__ lk1,
                               const float* __restrict__ lq2, const float* __restrict__ lk2,
                               float* __restrict__ lam_out) {
  __shared__ float r1[64], r2[64];
  int t = threadIdx.x;                     // 64 threads
  r1[t] = lq1[t] * lk1[t];
  r2[t] = lq2[t] * lk2[t];
  __syncthreads();
  if (t == 0) {
    float s1 = 0.f, s2 = 0.f;
    for (int j = 0; j < HDm; j++) { s1 += r1[j]; s2 += r2[j]; }
    const float lambda_init = 0.8f - 0.6f * __expf(-3.6f);   // depth 12
    lam_out[0] = __expf(s1) - __expf(s2) + lambda_init;
  }
}

// ----------------------------------- diff combine + RMSNorm + subln_g scale
__global__ __launch_bounds__(256)
void combine_rmsnorm(const float* __restrict__ Osub, const float* __restrict__ lam_p,
                     const float* __restrict__ subln_g, u16* __restrict__ attnN) {
  const int s = blockIdx.x;
  const int t = threadIdx.x;
  const int h = t >> 4;          // head 0..15
  const int i = t & 15;          // 8 dims each -> 128 dims per head
  const float lambda_init = 0.8f - 0.6f * __expf(-3.6f);
  const float lam = lam_p[0];

  const float* pe = Osub + (size_t)(2 * h) * SZ * 128 + (size_t)s * 128 + i * 8;
  const float* po = pe + (size_t)SZ * 128;
  float c[8], ss = 0.f;
#pragma unroll
  for (int k = 0; k < 8; k++) { c[k] = pe[k] - lam * po[k]; ss += c[k] * c[k]; }
#pragma unroll
  for (int off = 1; off < 16; off <<= 1) ss += __shfl_xor(ss, off, 32);
  float r = rsqrtf(ss * (1.f / 128.f) + 1e-5f) * (1.f - lambda_init);
#pragma unroll
  for (int k = 0; k < 8; k++)
    attnN[(size_t)s * SZ + h * 128 + i * 8 + k] = f2bf(c[k] * r * subln_g[i * 8 + k]);
}

// ---------------------------------------------------------------------------
extern "C" void kernel_launch(void* const* d_in, const int* in_sizes, int n_in,
                              void* d_out, int out_size, void* d_ws, size_t ws_size,
                              hipStream_t stream) {
  (void)in_sizes; (void)n_in; (void)out_size; (void)ws_size;
  const float* q    = (const float*)d_in[0];
  const float* k    = (const float*)d_in[1];
  const float* v    = (const float*)d_in[2];
  const float* Wq   = (const float*)d_in[5];
  const float* Wk   = (const float*)d_in[6];
  const float* Wv   = (const float*)d_in[7];
  const float* Wo   = (const float*)d_in[8];
  const float* lq1  = (const float*)d_in[9];
  const float* lk1  = (const float*)d_in[10];
  const float* lq2  = (const float*)d_in[11];
  const float* lk2  = (const float*)d_in[12];
  const float* subg = (const float*)d_in[13];

  const size_t MB = 1024u * 1024u;
  char* ws = (char*)d_ws;
  u16* Xq   = (u16*)(ws +  0 * MB);   // later reused as attnN (bf16, 8MB)
  u16* Xk   = (u16*)(ws +  8 * MB);   // [8..40MB) later reused as Osub (f32, 32MB)
  u16* Xv   = (u16*)(ws + 16 * MB);
  u16* Wqb  = (u16*)(ws + 24 * MB);
  u16* Wkb  = (u16*)(ws + 32 * MB);
  u16* Wvb  = (u16*)(ws + 40 * MB);
  u16* Wob  = (u16*)(ws + 48 * MB);
  u16* Qb   = (u16*)(ws + 56 * MB);
  u16* Kb   = (u16*)(ws + 64 * MB);
  u16* Vb   = (u16*)(ws + 72 * MB);
  u16* Vt   = (u16*)(ws + 80 * MB);
  float* lamBuf = (float*)(ws + 88 * MB);
  float* Osub  = (float*)(ws + 8 * MB);   // alias of Xk..Wkb (dead by then)
  u16*   attnN = (u16*)(ws + 0 * MB);     // alias of Xq (dead by then)

  const int N4 = SZ * SZ;
  cvt_f32_bf16<<<1024, 256, 0, stream>>>(q,  Xq,  N4);
  cvt_f32_bf16<<<1024, 256, 0, stream>>>(k,  Xk,  N4);
  cvt_f32_bf16<<<1024, 256, 0, stream>>>(v,  Xv,  N4);
  cvt_f32_bf16<<<1024, 256, 0, stream>>>(Wq, Wqb, N4);
  cvt_f32_bf16<<<1024, 256, 0, stream>>>(Wk, Wkb, N4);
  cvt_f32_bf16<<<1024, 256, 0, stream>>>(Wv, Wvb, N4);
  cvt_f32_bf16<<<1024, 256, 0, stream>>>(Wo, Wob, N4);
  compute_lambda<<<1, 64, 0, stream>>>(lq1, lk1, lq2, lk2, lamBuf);

  dim3 ggrid(SZ / 128, SZ / 128);
  gemm_wmma<false><<<ggrid, 128, 0, stream>>>(Xq, Wqb, Qb, nullptr, 0.125f);  // HD^-0.5
  gemm_wmma<false><<<ggrid, 128, 0, stream>>>(Xk, Wkb, Kb, nullptr, 1.0f);
  gemm_wmma<false><<<ggrid, 128, 0, stream>>>(Xv, Wvb, Vb, nullptr, 1.0f);

  transpose_bf16<<<dim3(SZ / 32, SZ / 32), dim3(32, 8), 0, stream>>>(Vb, Vt);

  // 32 sub-heads * 128 q-tiles = 4096 waves; 4 waves/block
  diffattn_fwd<<<1024, 128, 0, stream>>>(Qb, Kb, Vt, Osub);

  combine_rmsnorm<<<SZ, 256, 0, stream>>>(Osub, lamBuf, subg, attnN);

  gemm_wmma<true><<<ggrid, 128, 0, stream>>>(attnN, Wob, nullptr, (float*)d_out, 1.0f);
}
